// Generator_63015760167292
// MI455X (gfx1250) — matrix operand order repaired and checked by reading, not compiled, not run on hardware
//
#include <hip/hip_runtime.h>
#include <hip/hip_bf16.h>

#define B_  256
#define T_  256
#define H_  512
#define H4_ 2048

typedef __attribute__((ext_vector_type(16))) __bf16 bf16x16;
typedef __attribute__((ext_vector_type(8)))  float  v8f;
typedef __attribute__((ext_vector_type(4)))  int    v4i;

#define AS1 __attribute__((address_space(1)))
#define AS3 __attribute__((address_space(3)))

#if defined(__HIP_DEVICE_COMPILE__) && __has_builtin(__builtin_amdgcn_global_load_async_to_lds_b128)
#define HAVE_ASYNC_LDS 1
#else
#define HAVE_ASYNC_LDS 0
#endif

__device__ __forceinline__ void wait_async0() {
#if defined(__HIP_DEVICE_COMPILE__)
#if __has_builtin(__builtin_amdgcn_s_wait_asynccnt)
    __builtin_amdgcn_s_wait_asynccnt(0);
#else
    asm volatile("s_wait_asynccnt 0" ::: "memory");
#endif
#endif
}

__device__ __forceinline__ unsigned short f2bf(float x) {
    unsigned u; __builtin_memcpy(&u, &x, 4);
    unsigned r = (u + 0x7FFFu + ((u >> 16) & 1u)) >> 16;   // RNE
    return (unsigned short)r;
}
__device__ __forceinline__ float sigm(float x) { return 1.f / (1.f + __expf(-x)); }
__device__ __forceinline__ float lrelu(float x) { return fmaxf(x, 0.2f * x); }

// ---------------------------------------------------------------- utilities
__global__ void zero_kernel(uint4* p, int n4) {
    int i = blockIdx.x * blockDim.x + threadIdx.x;
    if (i < n4) p[i] = make_uint4(0, 0, 0, 0);
}

__global__ void state_init_kernel(const float* __restrict__ start, const float* __restrict__ endp,
                                  const float* __restrict__ init_in, const int* __restrict__ maxlen,
                                  float* __restrict__ pos, float* __restrict__ xs, float* __restrict__ outF) {
    int b = threadIdx.x;
    if (b < B_) {
        float s0 = start[2*b], s1 = start[2*b+1];
        float e0 = endp[2*b],  e1 = endp[2*b+1];
        pos[2*b] = s0; pos[2*b+1] = s1;
        float r0 = e0 - s0, r1 = e1 - s1;
        float rd = sqrtf(r0*r0 + r1*r1 + 1e-8f);
        float ra = atan2f(r1, r0);
        xs[b*8+0] = init_in[0]; xs[b*8+1] = init_in[1]; xs[b*8+2] = init_in[2];
        xs[b*8+3] = r0; xs[b*8+4] = r1; xs[b*8+5] = rd; xs[b*8+6] = ra; xs[b*8+7] = 0.f;
        ((int*)outF)[B_*T_*3 + b] = maxlen[0];
    }
}

// ---------------------------------------------------------------- condition encoder (one-time, fp32)
__global__ __launch_bounds__(512) void ce1_kernel(const float* __restrict__ start, const float* __restrict__ endp,
                                                  const float* __restrict__ z,
                                                  const float* __restrict__ w1, const float* __restrict__ b1,
                                                  const float* __restrict__ g1, const float* __restrict__ be1,
                                                  float* __restrict__ x1) {
    int b = blockIdx.x, n = threadIdx.x;
    float s0 = start[2*b], s1 = start[2*b+1], e0 = endp[2*b], e1 = endp[2*b+1];
    float dx = e0 - s0, dy = e1 - s1;
    float dist = sqrtf(dx*dx + dy*dy + 1e-8f);
    float ang  = atan2f(dy, dx);
    const float* wr = w1 + n*70;
    const float* zr = z + b*64;
    float acc = b1[n] + wr[0]*s0 + wr[1]*s1 + wr[2]*e0 + wr[3]*e1 + wr[4]*dist + wr[5]*ang;
    for (int k = 0; k < 64; ++k) acc += wr[6+k] * zr[k];
    __shared__ float sm[512];
    sm[n] = acc; __syncthreads();
    for (int st = 256; st > 0; st >>= 1) { if (n < st) sm[n] += sm[n+st]; __syncthreads(); }
    float mean = sm[0] * (1.f/512.f);
    __syncthreads();
    float d = acc - mean; sm[n] = d*d; __syncthreads();
    for (int st = 256; st > 0; st >>= 1) { if (n < st) sm[n] += sm[n+st]; __syncthreads(); }
    float var = sm[0] * (1.f/512.f);
    float y = d * rsqrtf(var + 1e-5f) * g1[n] + be1[n];
    x1[b*512 + n] = lrelu(y);
}

__global__ __launch_bounds__(512) void ce2_kernel(const float* __restrict__ x1,
                                                  const float* __restrict__ w2, const float* __restrict__ b2,
                                                  const float* __restrict__ g2, const float* __restrict__ be2,
                                                  float* __restrict__ condO) {
    int b = blockIdx.x, n = threadIdx.x;
    const float* wr = w2 + n*512;
    const float* xr = x1 + b*512;
    float acc = b2[n];
    for (int k = 0; k < 512; ++k) acc += wr[k] * xr[k];
    __shared__ float sm[512];
    sm[n] = acc; __syncthreads();
    for (int st = 256; st > 0; st >>= 1) { if (n < st) sm[n] += sm[n+st]; __syncthreads(); }
    float mean = sm[0] * (1.f/512.f);
    __syncthreads();
    float d = acc - mean; sm[n] = d*d; __syncthreads();
    for (int st = 256; st > 0; st >>= 1) { if (n < st) sm[n] += sm[n+st]; __syncthreads(); }
    float var = sm[0] * (1.f/512.f);
    float y = d * rsqrtf(var + 1e-5f) * g2[n] + be2[n];
    condO[b*512 + n] = lrelu(y);
}

// base0[b,n] = bih0[n]+bhh0[n] + cond[b,:] . Wih0[n, 3:515]   (time-invariant part hoisted out of the loop)
__global__ void base0_kernel(const float* __restrict__ cond, const float* __restrict__ Wih0,
                             const float* __restrict__ bih0, const float* __restrict__ bhh0,
                             float* __restrict__ base0) {
    int idx = blockIdx.x * 256 + threadIdx.x;
    int b = idx >> 11, n = idx & 2047;
    const float* wr = Wih0 + n*519 + 3;
    const float* cr = cond + b*512;
    float acc = bih0[n] + bhh0[n];
    for (int k = 0; k < 512; ++k) acc += cr[k] * wr[k];
    base0[idx] = acc;
}

// pack weights (N x K row-major fp32) -> bf16 WMMA B-tile layout:
// dst[((kt*NT + nt)*32 + lane)*16 + e] with n = nt*16+(lane&15), k = kt*32+(lane>>4)*16+e
__global__ void pack_b_kernel(unsigned short* __restrict__ dst,
                              const float* __restrict__ srcA, const float* __restrict__ srcB,
                              int k_split, int ld, int NT, int total) {
    int idx = blockIdx.x * 256 + threadIdx.x;
    if (idx >= total) return;
    int e = idx & 15, lane = (idx >> 4) & 31, tile = idx >> 9;
    int nt = tile % NT, kt = tile / NT;
    int n = nt*16 + (lane & 15);
    int k = kt*32 + (lane >> 4)*16 + e;
    float v = (k < k_split) ? srcA[n*ld + k] : srcB[n*ld + (k - k_split)];
    dst[idx] = f2bf(v);
}

__global__ void misc_kernel(const float* __restrict__ bih1, const float* __restrict__ bhh1,
                            const float* __restrict__ Wih0, float* __restrict__ bias1,
                            float* __restrict__ smallW) {
    int n = blockIdx.x * 256 + threadIdx.x;
    if (n >= H4_) return;
    bias1[n] = bih1[n] + bhh1[n];
    smallW[n*8+0] = Wih0[n*519 + 0];          // cur_in
    smallW[n*8+1] = Wih0[n*519 + 1];
    smallW[n*8+2] = Wih0[n*519 + 2];
    smallW[n*8+3] = Wih0[n*519 + 515];        // rem0
    smallW[n*8+4] = Wih0[n*519 + 516];        // rem1
    smallW[n*8+5] = Wih0[n*519 + 517];        // rd
    smallW[n*8+6] = Wih0[n*519 + 518];        // ra
    smallW[n*8+7] = 0.f;
}

// ---------------------------------------------------------------- per-step LSTM cell (WMMA GEMM + elementwise)
// grid 32 x 256: jt = bx>>1 (32-wide j slice of H), mt = (bx&1)*8 + wave (16-row batch tile)
__global__ __launch_bounds__(256) void lstm_kernel(
        const unsigned short* __restrict__ Alo, const unsigned short* __restrict__ Ahi,
        int kt_split, int KT,
        const unsigned short* __restrict__ Bp, int NT,
        const float* __restrict__ base, int mode,
        const float* __restrict__ smallW, const float* __restrict__ xs,
        float* __restrict__ c_state, unsigned short* __restrict__ hp_out,
        float* __restrict__ hf_out) {
    const int tid  = threadIdx.x;
    const int lane = tid & 31;
    const int jt   = blockIdx.x >> 1;
    const int mt   = ((blockIdx.x & 1) << 3) + (tid >> 5);

    int nt_s[8];
#pragma unroll
    for (int s = 0; s < 8; ++s) { int g = s >> 1, d = s & 1; nt_s[s] = g * (H_/16) + jt*2 + d; }

    v8f zero = {0.f,0.f,0.f,0.f,0.f,0.f,0.f,0.f};
    v8f acc[8];
#pragma unroll
    for (int s = 0; s < 8; ++s) acc[s] = zero;

    __shared__ uint4 sB4[512];                      // 8 B-tiles = 8KB, shared by all waves of the block
    const uint4* B4 = (const uint4*)Bp;

    for (int kt = 0; kt < KT; ++kt) {
        // stage 8 B tiles into LDS (async global->LDS direct path on gfx1250)
#if HAVE_ASYNC_LDS
        for (int f = tid; f < 512; f += 256) {
            int s = f >> 6, r64 = f & 63, ln_ = r64 >> 1, hf = f & 1;
            const uint4* src = &B4[((kt*NT + nt_s[s])*32 + ln_)*2 + hf];
            __builtin_amdgcn_global_load_async_to_lds_b128(
                (AS1 v4i*)src, (AS3 v4i*)&sB4[f], 0, 0);
        }
        wait_async0();
#else
        for (int f = tid; f < 512; f += 256) {
            int s = f >> 6, r64 = f & 63, ln_ = r64 >> 1, hf = f & 1;
            sB4[f] = B4[((kt*NT + nt_s[s])*32 + ln_)*2 + hf];
        }
#endif
        __syncthreads();
        const unsigned short* Asel = (kt < kt_split) ? Alo : Ahi;
        int ke = (kt < kt_split) ? kt : (kt - kt_split);
        bf16x16 a = ((const bf16x16*)Asel)[(ke*16 + mt)*32 + lane];
        const bf16x16* sBv = (const bf16x16*)sB4;
#pragma unroll
        for (int s = 0; s < 8; ++s) {
            bf16x16 bt = sBv[s*32 + lane];
            acc[s] = __builtin_amdgcn_wmma_f32_16x16x32_bf16(false, a, false, bt,
                                                             (short)0, acc[s], false, false);
        }
        __syncthreads();
    }

    const int hs = lane >> 4, ln = lane & 15;
#pragma unroll
    for (int r = 0; r < 8; ++r) {
        int b = mt*16 + r + 8*hs;
        float xq0=0,xq1=0,xq2=0,xq3=0,xq4=0,xq5=0,xq6=0;
        if (mode) {
            xq0 = xs[b*8+0]; xq1 = xs[b*8+1]; xq2 = xs[b*8+2];
            xq3 = xs[b*8+3]; xq4 = xs[b*8+4]; xq5 = xs[b*8+5]; xq6 = xs[b*8+6];
        }
#pragma unroll
        for (int d = 0; d < 2; ++d) {
            int j = jt*32 + d*16 + ln;
            float gate[4];
#pragma unroll
            for (int g = 0; g < 4; ++g) {
                int n = g*H_ + j;
                float bv;
                if (mode) {
                    bv = base[b*H4_ + n]
                       + xq0*smallW[n*8+0] + xq1*smallW[n*8+1] + xq2*smallW[n*8+2]
                       + xq3*smallW[n*8+3] + xq4*smallW[n*8+4] + xq5*smallW[n*8+5]
                       + xq6*smallW[n*8+6];
                } else {
                    bv = base[n];
                }
                gate[g] = acc[g*2 + d][r] + bv;
            }
            float iv = sigm(gate[0]);
            float fv = sigm(gate[1]);
            float gv = tanhf(gate[2]);
            float ov = sigm(gate[3]);
            float cold = c_state[b*H_ + j];
            float cnew = fv*cold + iv*gv;
            float hv = ov * tanhf(cnew);
            c_state[b*H_ + j] = cnew;
            // store h directly in packed WMMA A-tile layout (bf16)
            int lane_p = (d << 4) | (r + 8*hs);
            int e_p    = ln;
            hp_out[((jt*16 + mt)*32 + lane_p)*16 + e_p] = f2bf(hv);
            if (hf_out) hf_out[b*H_ + j] = hv;
        }
    }
}

// head GEMM: u = leaky( h2(256x512) . ow1^T(512x256) + ob1 )   via WMMA
__global__ __launch_bounds__(512) void head_gemm_kernel(const unsigned short* __restrict__ Ap,
                                                        const unsigned short* __restrict__ Bp,
                                                        const float* __restrict__ ob1,
                                                        float* __restrict__ u) {
    const int tid = threadIdx.x, lane = tid & 31, mt = tid >> 5, jt = blockIdx.x;
    v8f zero = {0.f,0.f,0.f,0.f,0.f,0.f,0.f,0.f};
    v8f acc0 = zero, acc1 = zero;
    __shared__ uint4 sB4[128];
    const uint4* B4 = (const uint4*)Bp;
    for (int kt = 0; kt < 16; ++kt) {
        if (tid < 128) {
            int s = tid >> 6, r64 = tid & 63, ln_ = r64 >> 1, hf = tid & 1;
            const uint4* src = &B4[((kt*16 + (jt*2 + s))*32 + ln_)*2 + hf];
#if HAVE_ASYNC_LDS
            __builtin_amdgcn_global_load_async_to_lds_b128(
                (AS1 v4i*)src, (AS3 v4i*)&sB4[tid], 0, 0);
#else
            sB4[tid] = *src;
#endif
        }
#if HAVE_ASYNC_LDS
        wait_async0();
#endif
        __syncthreads();
        bf16x16 a = ((const bf16x16*)Ap)[(kt*16 + mt)*32 + lane];
        const bf16x16* sBv = (const bf16x16*)sB4;
        acc0 = __builtin_amdgcn_wmma_f32_16x16x32_bf16(false, a, false, sBv[lane],      (short)0, acc0, false, false);
        acc1 = __builtin_amdgcn_wmma_f32_16x16x32_bf16(false, a, false, sBv[32 + lane], (short)0, acc1, false, false);
        __syncthreads();
    }
    const int hs = lane >> 4, ln = lane & 15;
#pragma unroll
    for (int r = 0; r < 8; ++r) {
        int b = mt*16 + r + 8*hs;
#pragma unroll
        for (int d = 0; d < 2; ++d) {
            int n = (jt*2 + d)*16 + ln;
            float v = (d ? acc1[r] : acc0[r]) + ob1[n];
            u[b*256 + n] = lrelu(v);
        }
    }
}

// head part 2: o = u . ow2^T + ob2, softplus, pos update, next-step inputs
__global__ void head2_kernel(const float* __restrict__ u, const float* __restrict__ ow2,
                             const float* __restrict__ ob2, const float* __restrict__ endp,
                             float* __restrict__ pos, float* __restrict__ xs,
                             float* __restrict__ outF, int t) {
    int b = threadIdx.x;
    float a0 = ob2[0], a1 = ob2[1], a2 = ob2[2];
    const float* ur = u + b*256;
    for (int k = 0; k < 256; ++k) {
        float uv = ur[k];
        a0 += uv * ow2[k];
        a1 += uv * ow2[256 + k];
        a2 += uv * ow2[512 + k];
    }
    float sp   = (a2 > 20.f) ? a2 : log1pf(__expf(a2));
    float out2 = sp + 0.001f;
    outF[(b*T_ + t)*3 + 0] = a0;
    outF[(b*T_ + t)*3 + 1] = a1;
    outF[(b*T_ + t)*3 + 2] = out2;
    float p0 = pos[2*b]   + a0;
    float p1 = pos[2*b+1] + a1;
    pos[2*b] = p0; pos[2*b+1] = p1;
    float r0 = endp[2*b] - p0, r1 = endp[2*b+1] - p1;
    float rd = sqrtf(r0*r0 + r1*r1 + 1e-8f);
    float ra = atan2f(r1, r0);
    xs[b*8+0] = a0; xs[b*8+1] = a1; xs[b*8+2] = out2;
    xs[b*8+3] = r0; xs[b*8+4] = r1; xs[b*8+5] = rd; xs[b*8+6] = ra;
}

// ---------------------------------------------------------------- host
extern "C" void kernel_launch(void* const* d_in, const int* in_sizes, int n_in,
                              void* d_out, int out_size, void* d_ws, size_t ws_size,
                              hipStream_t stream) {
    (void)in_sizes; (void)n_in; (void)out_size; (void)ws_size;
    const float* start  = (const float*)d_in[0];
    const float* endp   = (const float*)d_in[1];
    const float* z      = (const float*)d_in[2];
    const float* ce_w1  = (const float*)d_in[3];
    const float* ce_b1  = (const float*)d_in[4];
    const float* ce_g1  = (const float*)d_in[5];
    const float* ce_be1 = (const float*)d_in[6];
    const float* ce_w2  = (const float*)d_in[7];
    const float* ce_b2  = (const float*)d_in[8];
    const float* ce_g2  = (const float*)d_in[9];
    const float* ce_be2 = (const float*)d_in[10];
    const float* Wih0   = (const float*)d_in[11];
    const float* Whh0   = (const float*)d_in[12];
    const float* bih0   = (const float*)d_in[13];
    const float* bhh0   = (const float*)d_in[14];
    const float* Wih1   = (const float*)d_in[15];
    const float* Whh1   = (const float*)d_in[16];
    const float* bih1   = (const float*)d_in[17];
    const float* bhh1   = (const float*)d_in[18];
    const float* ow1    = (const float*)d_in[19];
    const float* ob1    = (const float*)d_in[20];
    const float* ow2    = (const float*)d_in[21];
    const float* ob2    = (const float*)d_in[22];
    const float* init_in= (const float*)d_in[23];
    const int*   maxlen = (const int*)d_in[24];
    float* outF = (float*)d_out;

    char* w = (char*)d_ws; size_t off = 0;
    auto alloc = [&](size_t bytes) -> void* {
        void* p = w + off; off = (off + bytes + 255) & ~(size_t)255; return p;
    };
    float* cond   = (float*)alloc((size_t)B_*H_*4);
    float* x1t    = (float*)alloc((size_t)B_*H_*4);
    float* base0  = (float*)alloc((size_t)B_*H4_*4);
    float* bias1  = (float*)alloc((size_t)H4_*4);
    float* smallW = (float*)alloc((size_t)H4_*8*4);
    unsigned short* Whh0p = (unsigned short*)alloc((size_t)16*128*512*2);   // 2 MB
    unsigned short* W1p   = (unsigned short*)alloc((size_t)32*128*512*2);   // 4 MB (concat [Wih1;Whh1])
    unsigned short* ow1p  = (unsigned short*)alloc((size_t)16*16*512*2);
    unsigned short* h1pA  = (unsigned short*)alloc((size_t)16*16*512*2);    // packed h, double buffered
    unsigned short* h1pB  = (unsigned short*)alloc((size_t)16*16*512*2);
    unsigned short* h2pA  = (unsigned short*)alloc((size_t)16*16*512*2);
    unsigned short* h2pB  = (unsigned short*)alloc((size_t)16*16*512*2);
    float* c1  = (float*)alloc((size_t)B_*H_*4);
    float* c2  = (float*)alloc((size_t)B_*H_*4);
    float* u   = (float*)alloc((size_t)B_*256*4);
    float* xs  = (float*)alloc((size_t)B_*8*4);
    float* pos = (float*)alloc((size_t)B_*2*4);

    // one-time setup (every call, deterministic)
    zero_kernel<<<64, 256, 0, stream>>>((uint4*)h1pA, 16384);
    zero_kernel<<<64, 256, 0, stream>>>((uint4*)h2pA, 16384);
    zero_kernel<<<128, 256, 0, stream>>>((uint4*)c1, 32768);
    zero_kernel<<<128, 256, 0, stream>>>((uint4*)c2, 32768);
    state_init_kernel<<<1, 256, 0, stream>>>(start, endp, init_in, maxlen, pos, xs, outF);
    ce1_kernel<<<B_, 512, 0, stream>>>(start, endp, z, ce_w1, ce_b1, ce_g1, ce_be1, x1t);
    ce2_kernel<<<B_, 512, 0, stream>>>(x1t, ce_w2, ce_b2, ce_g2, ce_be2, cond);
    base0_kernel<<<(B_*H4_)/256, 256, 0, stream>>>(cond, Wih0, bih0, bhh0, base0);
    pack_b_kernel<<<(16*128*512)/256, 256, 0, stream>>>(Whh0p, Whh0, Whh0, 512, 512, 128, 16*128*512);
    pack_b_kernel<<<(32*128*512)/256, 256, 0, stream>>>(W1p, Wih1, Whh1, 512, 512, 128, 32*128*512);
    pack_b_kernel<<<(16*16*512)/256, 256, 0, stream>>>(ow1p, ow1, ow1, 512, 512, 16, 16*16*512);
    misc_kernel<<<8, 256, 0, stream>>>(bih1, bhh1, Wih0, bias1, smallW);

    unsigned short* h1p[2] = {h1pA, h1pB};
    unsigned short* h2p[2] = {h2pA, h2pB};
    for (int t = 0; t < T_; ++t) {
        int cur = t & 1, nxt = (t + 1) & 1;
        // layer 0: gates = base0 + smallW.xs + h1_prev . Whh0^T   (K=512)
        lstm_kernel<<<32, 256, 0, stream>>>(h1p[cur], h1p[cur], 16, 16, Whh0p, 128,
                                            base0, 1, smallW, xs, c1, h1p[nxt], (float*)nullptr);
        // layer 1: gates = bias1 + [h1_t ; h2_prev] . [Wih1;Whh1]^T   (K=1024)
        lstm_kernel<<<32, 256, 0, stream>>>(h1p[nxt], h2p[cur], 16, 32, W1p, 128,
                                            bias1, 0, smallW, xs, c2, h2p[nxt], (float*)nullptr);
        head_gemm_kernel<<<8, 512, 0, stream>>>(h2p[nxt], ow1p, ob1, u);
        head2_kernel<<<1, 256, 0, stream>>>(u, ow2, ob2, endp, pos, xs, outF, t);
    }
}
